// QuantAttentionBlock_11355893530741
// MI455X (gfx1250) — matrix operand order repaired and checked by reading, not compiled, not run on hardware
//
#include <hip/hip_runtime.h>
#include <hip/hip_bf16.h>
#include <stdint.h>
#include <math.h>

// ---------------------------------------------------------------------------
// QuantAttentionBlock for MI455X (gfx1250, wave32, WMMA f16 16x16x32).
// Pipeline: init -> wcvt -> gn_stats -> gn_apply -> qkv WMMA gemm (+minmax)
//           -> quant params -> flash attention (WMMA) -> proj WMMA gemm.
// GEMMs use CDNA5 async global->LDS copies (ASYNCcnt) with double buffering.
// ---------------------------------------------------------------------------

typedef __attribute__((ext_vector_type(16))) _Float16 v16h;
typedef __attribute__((ext_vector_type(8)))  _Float16 v8h;
typedef __attribute__((ext_vector_type(4)))  _Float16 v4h;
typedef __attribute__((ext_vector_type(8)))  float    v8f;
typedef __attribute__((ext_vector_type(4)))  float    v4f;

#define DEV __device__ __forceinline__

DEV v8f wmma16(v16h a, v16h b, v8f c) {
  // D = A(16x32 f16) * B(32x16 f16) + C(16x16 f32)
  return __builtin_amdgcn_wmma_f32_16x16x32_f16(false, a, false, b, (short)0, c,
                                                false, false);
}

DEV v8f vzero8() {
  v8f z;
#pragma unroll
  for (int i = 0; i < 8; i++) z[i] = 0.0f;
  return z;
}

// strided gather -> B fragment (lane = N column, elem e = K row e + 8*(e>>3) + 8*hg)
DEV v16h ld16(const _Float16* p, int ld, int hg) {
  v16h o;
#pragma unroll
  for (int i = 0; i < 8; i++) { o[i] = p[(8 * hg + i) * ld]; o[i + 8] = p[(16 + 8 * hg + i) * ld]; }
  return o;
}

// A fragment: source row (row-major, K contiguous), 32-wide K window at colbase.
// element e holds K = colbase + e + 8*(e>>3) + 8*hg  (per ISA 16-bit A layout)
DEV v16h ld_afrag(const _Float16* row, int colbase, int hg) {
  v8h lo = *(const v8h*)(row + colbase + 8 * hg);
  v8h hi = *(const v8h*)(row + colbase + 16 + 8 * hg);
  v16h o;
#pragma unroll
  for (int i = 0; i < 8; i++) { o[i] = lo[i]; o[i + 8] = hi[i]; }
  return o;
}

// --- CDNA5 async global->LDS copy (VGLOBAL encoding, tracked by ASYNCcnt) ---
// Low 32 bits of a generic pointer into __shared__ are the wave-relative LDS
// address (ISA 10.2 aperture rules), which is exactly what VDST wants.
DEV uint32_t lds_off(const void* p) { return (uint32_t)(uintptr_t)p; }

DEV void async_ld_b128(uint32_t lds_addr, const void* gaddr) {
  asm volatile("global_load_async_to_lds_b128 %0, %1, off"
               :: "v"(lds_addr), "v"(gaddr)
               : "memory");
}
DEV void wait_async0() {
  asm volatile("s_wait_asynccnt 0x0" ::: "memory");
}

// monotone float <-> uint32 mapping for min/max atomics
DEV uint32_t fenc(float x) {
  uint32_t u = __float_as_uint(x);
  return (u & 0x80000000u) ? ~u : (u | 0x80000000u);
}
DEV float fdec(uint32_t e) {
  uint32_t u = (e & 0x80000000u) ? (e & 0x7fffffffu) : ~e;
  return __uint_as_float(u);
}

// UniformAffineQuantizer fake-quant (8-bit, asymmetric), RNE rounding like jnp.round
DEV float fquant(float x, float s, float zp) {
  float q = rintf(x / s) + zp;
  q = fminf(fmaxf(q, 0.0f), 255.0f);
  return (q - zp) * s;
}

// ---------------------------------------------------------------------------
__global__ void k_init(uint32_t* mm) {
  int i = threadIdx.x;
  if (i < 3) mm[2 * i] = 0xFFFFFFFFu;        // running-min slots (encoded +inf)
  else if (i < 6) mm[2 * (i - 3) + 1] = 0u;  // running-max slots (encoded -inf)
}

__global__ __launch_bounds__(256) void k_wcvt(const float* __restrict__ qkvw,
                                              const float* __restrict__ projw,
                                              _Float16* __restrict__ wq,
                                              _Float16* __restrict__ wp) {
  int i = blockIdx.x * 256 + threadIdx.x;  // exactly 786432 + 262144 threads
  if (i < 786432) wq[i] = (_Float16)qkvw[i];
  else            wp[i - 786432] = (_Float16)projw[i - 786432];
}

// ---------------------------------------------------------------------------
// GroupNorm: stats per (b, group): 16 channels x 1024 = 16384 elements
__global__ __launch_bounds__(256) void k_gnstats(const float* __restrict__ x,
                                                 float* __restrict__ stats) {
  int bg = blockIdx.x;  // b*32+g ; channel base offset = bg*16*1024
  const float* base = x + (size_t)bg * 16384;
  float s = 0.f, s2 = 0.f;
  for (int i = threadIdx.x; i < 16384; i += 256) {
    float v = base[i];
    s += v; s2 += v * v;
  }
  __shared__ float r1[256], r2[256];
  r1[threadIdx.x] = s; r2[threadIdx.x] = s2;
  __syncthreads();
  for (int off = 128; off > 0; off >>= 1) {
    if (threadIdx.x < off) {
      r1[threadIdx.x] += r1[threadIdx.x + off];
      r2[threadIdx.x] += r2[threadIdx.x + off];
    }
    __syncthreads();
  }
  if (threadIdx.x == 0) {
    float mean = r1[0] * (1.0f / 16384.0f);
    float var  = r2[0] * (1.0f / 16384.0f) - mean * mean;
    stats[2 * bg] = mean;
    stats[2 * bg + 1] = rsqrtf(var + 1e-5f);
  }
}

__global__ __launch_bounds__(256) void k_gnapply(const float* __restrict__ x,
                                                 const float* __restrict__ stats,
                                                 const float* __restrict__ w,
                                                 const float* __restrict__ bgn,
                                                 _Float16* __restrict__ h) {
  size_t i = ((size_t)blockIdx.x * 256 + threadIdx.x) * 4;
  int c = (int)((i >> 10) & 511);
  int b = (int)(i >> 19);
  int bg = b * 32 + (c >> 4);
  float mean = stats[2 * bg], rstd = stats[2 * bg + 1];
  float sw = w[c] * rstd;
  float sb = bgn[c] - mean * sw;
  v4f xv = *(const v4f*)(x + i);
  v4h hv;
#pragma unroll
  for (int j = 0; j < 4; j++) hv[j] = (_Float16)(xv[j] * sw + sb);
  *(v4h*)(h + i) = hv;
}

// ---------------------------------------------------------------------------
// QKV GEMM: D(o,t) = W(1536x512) * H(512x1024) + bias  per batch.
// Block tile 128x128, K-chunk 32, double-buffered async LDS fills.
// 8 waves; wave tile = 32(M) x 64(N). Epilogue splits into q/k/v tensors laid
// out (bh, t, c) f16 and reduces global min/max for the fake-quant params.
__global__ __launch_bounds__(256) void k_qkv(const _Float16* __restrict__ W,
                                             const _Float16* __restrict__ H,
                                             const float* __restrict__ bias,
                                             _Float16* __restrict__ q,
                                             _Float16* __restrict__ k,
                                             _Float16* __restrict__ v,
                                             uint32_t* __restrict__ mm) {
  __shared__ _Float16 As[2][128 * 32];    // (m, k) row-major, lda=32
  __shared__ _Float16 Bs[2][32 * 136];    // (k, n) row-major, ldb=136 (pad)
  int b = blockIdx.z;
  int m0 = blockIdx.y * 128, n0 = blockIdx.x * 128;
  int tid = threadIdx.x, lane = tid & 31, wave = tid >> 5;
  int l15 = lane & 15, hg = lane >> 4;
  int wm = wave & 3, wn = wave >> 2;

  v8f acc[2][4];
#pragma unroll
  for (int a1 = 0; a1 < 2; a1++)
#pragma unroll
    for (int a2 = 0; a2 < 4; a2++) acc[a1][a2] = vzero8();

  int ar = tid >> 1, ac = (tid & 1) * 16;
  int br = tid >> 3, bc = (tid & 7) * 16;
  const _Float16* arow = W + (size_t)(m0 + ar) * 512 + ac;          // + kc
  const _Float16* brow = H + ((size_t)b * 512 + br) * 1024 + n0 + bc;  // + kc*1024

  // prologue: chunk 0 -> buffer 0 (async engine, no VGPR staging)
  async_ld_b128(lds_off(&As[0][ar * 32 + ac]),     arow);
  async_ld_b128(lds_off(&As[0][ar * 32 + ac + 8]), arow + 8);
  async_ld_b128(lds_off(&Bs[0][br * 136 + bc]),     brow);
  async_ld_b128(lds_off(&Bs[0][br * 136 + bc + 8]), brow + 8);
  wait_async0();
  __syncthreads();

  int cur = 0;
  for (int kc = 0; kc < 512; kc += 32) {
    if (kc + 32 < 512) {  // issue next chunk into the other buffer
      int nb = cur ^ 1;
      const _Float16* an = arow + kc + 32;
      const _Float16* bn = brow + (size_t)(kc + 32) * 1024;
      async_ld_b128(lds_off(&As[nb][ar * 32 + ac]),     an);
      async_ld_b128(lds_off(&As[nb][ar * 32 + ac + 8]), an + 8);
      async_ld_b128(lds_off(&Bs[nb][br * 136 + bc]),     bn);
      async_ld_b128(lds_off(&Bs[nb][br * 136 + bc + 8]), bn + 8);
    }
    v16h af0 = ld_afrag(&As[cur][(wm * 32 + l15) * 32], 0, hg);
    v16h af1 = ld_afrag(&As[cur][(wm * 32 + 16 + l15) * 32], 0, hg);
    const _Float16* bsr = &Bs[cur][l15 + wn * 64];
#pragma unroll
    for (int jn = 0; jn < 4; jn++) {
      v16h bf = ld16(bsr + jn * 16, 136, hg);
      acc[0][jn] = wmma16(af0, bf, acc[0][jn]);
      acc[1][jn] = wmma16(af1, bf, acc[1][jn]);
    }
    wait_async0();
    __syncthreads();
    cur ^= 1;
  }

  // epilogue: o in [m0+wm*32, +32) lies in one (head, q/k/v) band (32 | 64 | 192)
  int o_base = m0 + wm * 32;
  int head = o_base / 192;
  int sel = (o_base % 192) / 64;  // 0=q 1=k 2=v
  _Float16* dst = (sel == 0) ? q : ((sel == 1) ? k : v);
  float vmin = INFINITY, vmax = -INFINITY;
#pragma unroll
  for (int im = 0; im < 2; im++) {
    int olo = o_base + im * 16 + 8 * hg;  // +r gives this lane's o
    int c0 = olo & 63;                    // channel within head (contig over r)
#pragma unroll
    for (int jn = 0; jn < 4; jn++) {
      int t = n0 + wn * 64 + jn * 16 + l15;
      v8h pk;
#pragma unroll
      for (int r = 0; r < 8; r++) {
        float val = acc[im][jn][r] + bias[olo + r];
        vmin = fminf(vmin, val);
        vmax = fmaxf(vmax, val);
        pk[r] = (_Float16)val;
      }
      *(v8h*)(dst + (((size_t)(b * 8 + head) * 1024 + t) << 6) + c0) = pk;
    }
  }
#pragma unroll
  for (int msk = 1; msk < 32; msk <<= 1) {
    vmin = fminf(vmin, __shfl_xor(vmin, msk, 32));
    vmax = fmaxf(vmax, __shfl_xor(vmax, msk, 32));
  }
  if (lane == 0) {
    atomicMin(&mm[2 * sel], fenc(vmin));
    atomicMax(&mm[2 * sel + 1], fenc(vmax));
  }
}

__global__ void k_qparams(const uint32_t* __restrict__ mm, float* __restrict__ qp) {
  int i = threadIdx.x;
  if (i < 3) {
    float mn = fminf(fdec(mm[2 * i]), 0.0f);
    float mx = fmaxf(fdec(mm[2 * i + 1]), 0.0f);
    float s = fmaxf((mx - mn) * (1.0f / 255.0f), 1e-8f);
    qp[2 * i] = s;
    qp[2 * i + 1] = rintf(-mn / s);
  }
}

// ---------------------------------------------------------------------------
// Flash attention. Block = (bh, 128-query tile); wave owns 16 queries.
// q/k/v global layout: (bh, t, c) f16 (raw); fake-quant applied on load.
__global__ __launch_bounds__(256) void k_attn(const _Float16* __restrict__ Q,
                                              const _Float16* __restrict__ K,
                                              const _Float16* __restrict__ V,
                                              const float* __restrict__ qp,
                                              _Float16* __restrict__ A) {
  __shared__ _Float16 Ks[64 * 72];       // (c, s) -- B operand for S = Q^T K
  __shared__ _Float16 Vs[64 * 72];       // (s, c) -- B operand for O = P V^T
  __shared__ _Float16 Ps[8 * 16 * 72];   // per-wave P re-layout scratch
  int bh = blockIdx.x, qb = blockIdx.y;
  int tid = threadIdx.x, lane = tid & 31, wave = tid >> 5;
  int l15 = lane & 15, hg = lane >> 4;
  float sq = qp[0], zq = qp[1], sk = qp[2], zk = qp[3], sv = qp[4], zv = qp[5];
  const float ascale = 0.35355339059327373f;  // 1/sqrt(sqrt(64))

  int t0 = qb * 128 + wave * 16;
  const _Float16* qrow = Q + ((size_t)bh * 1024 + t0 + l15) * 64;
  v16h aq[2];
#pragma unroll
  for (int kk = 0; kk < 2; kk++) {
    v16h raw = ld_afrag(qrow, kk * 32, hg);
    v16h qz;
#pragma unroll
    for (int e = 0; e < 16; e++)
      qz[e] = (_Float16)(fquant((float)raw[e], sq, zq) * ascale);
    aq[kk] = qz;
  }

  v8f O[4];
  float mr[8], lr[8];
#pragma unroll
  for (int j = 0; j < 4; j++) O[j] = vzero8();
#pragma unroll
  for (int r = 0; r < 8; r++) { mr[r] = -INFINITY; lr[r] = 0.0f; }
  _Float16* pw = Ps + wave * 16 * 72;

  for (int s0 = 0; s0 < 1024; s0 += 64) {
    // cooperative fill: K chunk transposed to (c,s), V chunk direct (s,c)
#pragma unroll
    for (int it = 0; it < 16; it++) {
      int idx = tid * 16 + it;
      int s = idx >> 6, c = idx & 63;
      size_t g = ((size_t)bh * 1024 + s0 + s) * 64 + c;
      Ks[c * 72 + s] = (_Float16)(fquant((float)K[g], sk, zk) * ascale);
      Vs[s * 72 + c] = (_Float16)(fquant((float)V[g], sv, zv));
    }
    __syncthreads();

    // S(16 x 64) = Q^T * K-chunk (two K=32 steps over c)
    v8f S[4];
#pragma unroll
    for (int j = 0; j < 4; j++) {
      v8f sa = vzero8();
      sa = wmma16(aq[0], ld16(Ks + l15 + j * 16, 72, hg), sa);
      sa = wmma16(aq[1], ld16(Ks + 32 * 72 + l15 + j * 16, 72, hg), sa);
      S[j] = sa;
    }

    // online softmax (row = t, spread across 16 lanes of each half-group)
    float alpha[8];
#pragma unroll
    for (int r = 0; r < 8; r++) {
      float mv = fmaxf(fmaxf(S[0][r], S[1][r]), fmaxf(S[2][r], S[3][r]));
#pragma unroll
      for (int msk = 1; msk < 16; msk <<= 1)
        mv = fmaxf(mv, __shfl_xor(mv, msk, 32));
      float mn = fmaxf(mr[r], mv);
      alpha[r] = __expf(mr[r] - mn);
      mr[r] = mn;
      float rs = 0.0f;
#pragma unroll
      for (int j = 0; j < 4; j++) {
        float p = __expf(S[j][r] - mn);
        S[j][r] = p;
        rs += p;
      }
#pragma unroll
      for (int msk = 1; msk < 16; msk <<= 1) rs += __shfl_xor(rs, msk, 32);
      lr[r] = lr[r] * alpha[r] + rs;
    }
#pragma unroll
    for (int j = 0; j < 4; j++)
#pragma unroll
      for (int r = 0; r < 8; r++) O[j][r] *= alpha[r];

    // re-layout P: C-layout -> (t, s) rows in LDS, then read back as A frags
#pragma unroll
    for (int j = 0; j < 4; j++)
#pragma unroll
      for (int r = 0; r < 8; r++)
        pw[(r + 8 * hg) * 72 + j * 16 + l15] = (_Float16)S[j][r];

#pragma unroll
    for (int kk = 0; kk < 2; kk++) {
      v16h pf = ld_afrag(pw + l15 * 72, kk * 32, hg);
#pragma unroll
      for (int j = 0; j < 4; j++) {
        v16h vf = ld16(Vs + kk * 32 * 72 + l15 + j * 16, 72, hg);
        O[j] = wmma16(pf, vf, O[j]);
      }
    }
    __syncthreads();
  }

  // normalize and store to a: (b, C, t) f16, row = bh*64 + c
#pragma unroll
  for (int j = 0; j < 4; j++) {
    v8h pk;
#pragma unroll
    for (int r = 0; r < 8; r++) pk[r] = (_Float16)(O[j][r] / lr[r]);
    int c = j * 16 + l15;
    *(v8h*)(A + ((size_t)bh * 64 + c) * 1024 + t0 + 8 * hg) = pk;
  }
}

// ---------------------------------------------------------------------------
// Proj GEMM: out = proj_w(512x512) * a(512x1024) + bias + x   (f32 out)
__global__ __launch_bounds__(256) void k_proj(const _Float16* __restrict__ W,
                                              const _Float16* __restrict__ Ain,
                                              const float* __restrict__ bias,
                                              const float* __restrict__ X,
                                              float* __restrict__ out) {
  __shared__ _Float16 As[2][128 * 32];
  __shared__ _Float16 Bs[2][32 * 136];
  int b = blockIdx.z;
  int m0 = blockIdx.y * 128, n0 = blockIdx.x * 128;
  int tid = threadIdx.x, lane = tid & 31, wave = tid >> 5;
  int l15 = lane & 15, hg = lane >> 4;
  int wm = wave & 3, wn = wave >> 2;

  v8f acc[2][4];
#pragma unroll
  for (int a1 = 0; a1 < 2; a1++)
#pragma unroll
    for (int a2 = 0; a2 < 4; a2++) acc[a1][a2] = vzero8();

  int ar = tid >> 1, ac = (tid & 1) * 16;
  int br = tid >> 3, bc = (tid & 7) * 16;
  const _Float16* arow = W + (size_t)(m0 + ar) * 512 + ac;
  const _Float16* brow = Ain + ((size_t)b * 512 + br) * 1024 + n0 + bc;

  async_ld_b128(lds_off(&As[0][ar * 32 + ac]),     arow);
  async_ld_b128(lds_off(&As[0][ar * 32 + ac + 8]), arow + 8);
  async_ld_b128(lds_off(&Bs[0][br * 136 + bc]),     brow);
  async_ld_b128(lds_off(&Bs[0][br * 136 + bc + 8]), brow + 8);
  wait_async0();
  __syncthreads();

  int cur = 0;
  for (int kc = 0; kc < 512; kc += 32) {
    if (kc + 32 < 512) {
      int nb = cur ^ 1;
      const _Float16* an = arow + kc + 32;
      const _Float16* bn = brow + (size_t)(kc + 32) * 1024;
      async_ld_b128(lds_off(&As[nb][ar * 32 + ac]),     an);
      async_ld_b128(lds_off(&As[nb][ar * 32 + ac + 8]), an + 8);
      async_ld_b128(lds_off(&Bs[nb][br * 136 + bc]),     bn);
      async_ld_b128(lds_off(&Bs[nb][br * 136 + bc + 8]), bn + 8);
    }
    v16h af0 = ld_afrag(&As[cur][(wm * 32 + l15) * 32], 0, hg);
    v16h af1 = ld_afrag(&As[cur][(wm * 32 + 16 + l15) * 32], 0, hg);
    const _Float16* bsr = &Bs[cur][l15 + wn * 64];
#pragma unroll
    for (int jn = 0; jn < 4; jn++) {
      v16h bf = ld16(bsr + jn * 16, 136, hg);
      acc[0][jn] = wmma16(af0, bf, acc[0][jn]);
      acc[1][jn] = wmma16(af1, bf, acc[1][jn]);
    }
    wait_async0();
    __syncthreads();
    cur ^= 1;
  }

  int o_base = m0 + wm * 32;
#pragma unroll
  for (int im = 0; im < 2; im++) {
#pragma unroll
    for (int jn = 0; jn < 4; jn++) {
      int t = n0 + wn * 64 + jn * 16 + l15;
#pragma unroll
      for (int r = 0; r < 8; r++) {
        int o = o_base + im * 16 + 8 * hg + r;
        size_t idx = ((size_t)b * 512 + o) * 1024 + t;
        out[idx] = acc[im][jn][r] + bias[o] + X[idx];
      }
    }
  }
}

// ---------------------------------------------------------------------------
extern "C" void kernel_launch(void* const* d_in, const int* in_sizes, int n_in,
                              void* d_out, int out_size, void* d_ws, size_t ws_size,
                              hipStream_t stream) {
  const float* x      = (const float*)d_in[0];
  const float* gn_w   = (const float*)d_in[1];
  const float* gn_b   = (const float*)d_in[2];
  const float* qkv_w  = (const float*)d_in[3];
  const float* qkv_b  = (const float*)d_in[4];
  const float* proj_w = (const float*)d_in[5];
  const float* proj_b = (const float*)d_in[6];
  float* out = (float*)d_out;

  uint8_t* ws = (uint8_t*)d_ws;
  uint32_t* mm   = (uint32_t*)ws;              // 6 u32 (encoded min/max q,k,v)
  float*    qp   = (float*)(ws + 64);          // 6 f32 (scale, zp) x3
  float*    st   = (float*)(ws + 256);         // 512 f32 groupnorm stats
  _Float16* wq   = (_Float16*)(ws + 4096);     // 1536x512
  _Float16* wp   = wq + 786432;                // 512x512
  _Float16* h    = wp + 262144;                // (B,C,T) f16
  _Float16* qh   = h + 4194304;                // (bh,t,c) f16
  _Float16* kh   = qh + 4194304;
  _Float16* vh   = kh + 4194304;
  _Float16* ah   = h;                          // alias: h dead after qkv gemm
  (void)in_sizes; (void)n_in; (void)out_size; (void)ws_size;

  k_init<<<1, 64, 0, stream>>>(mm);
  k_wcvt<<<4096, 256, 0, stream>>>(qkv_w, proj_w, wq, wp);
  k_gnstats<<<256, 256, 0, stream>>>(x, st);
  k_gnapply<<<4096, 256, 0, stream>>>(x, st, gn_w, gn_b, h);
  k_qkv<<<dim3(8, 12, 8), 256, 0, stream>>>(wq, h, qkv_b, qh, kh, vh, mm);
  k_qparams<<<1, 32, 0, stream>>>(mm, qp);
  k_attn<<<dim3(64, 8), 256, 0, stream>>>(qh, kh, vh, qp, ah);
  k_proj<<<dim3(8, 4, 8), 256, 0, stream>>>(wp, ah, proj_b, x, out);
}
